// CSBrainLLMVQ_3753801417216
// MI455X (gfx1250) — matrix operand order repaired and checked by reading, not compiled, not run on hardware
//
#include <hip/hip_runtime.h>
#include <hip/hip_bf16.h>

// ---------------------------------------------------------------------------
// CSBrain LLM-VQ forward on MI455X (gfx1250, wave32, WMMA).
// Dominant cost: 18240x4096 (K=4096) token-vs-codebook score GEMM (~612 GFLOP)
// -> v_wmma_f32_16x16x32_bf16, codebook bf16 (33.5 MB, L2-resident), argmin
// fused into the GEMM epilogue. The 16-token A-tile is staged once per block
// into LDS (128 KB of the 320 KB WGP LDS) and each LDS A-fragment feeds 4
// WMMAs (4 code tiles register-blocked) to cut L2/LDS traffic.
// ---------------------------------------------------------------------------

#define B_    32
#define CH_   19
#define NP_P  30
#define PS_   200
#define DM_   200
#define LLM_  4096
#define KC_   4096
#define T_    (CH_ * NP_P)        // 570 tokens per batch
#define NT_   (B_ * T_)           // 18240 tokens total
#define KP_   224                 // DM padded to multiple of 32 for WMMA K
#define NO_   208                 // DM padded to multiple of 16 for N tiles
#define NF_   101                 // rfft bins (PS/2+1)
#define ASTRIDE_ (LLM_ + 16)      // LDS A row stride (elems), pad breaks bank dup

typedef __attribute__((ext_vector_type(16))) __bf16 v16bf;
typedef __attribute__((ext_vector_type(8)))  float  v8f;

union FragU { uint4 q[2]; v16bf v; };

// A fragment 16x32 bf16: lane = M row (lane&15); VGPR0-3 hold K {0..7|8..15},
// VGPR4-7 hold K {16..23|24..31} depending on lane half. Two 16B loads.
__device__ __forceinline__ v16bf load_a_frag(const __bf16* base, int stride,
                                             int row, int kbase, int lane) {
  const __bf16* p = base + (size_t)row * stride + kbase + ((lane >> 4) << 3);
  FragU f;
  f.q[0] = *(const uint4*)p;
  f.q[1] = *(const uint4*)(p + 16);
  return f.v;
}

// Same A fragment sourced from an LDS-staged tile (ds_load_b128 x2).
__device__ __forceinline__ v16bf lds_a_frag(const __bf16* sA, int row, int kbase,
                                            int lane) {
  const __bf16* p = sA + row * ASTRIDE_ + kbase + ((lane >> 4) << 3);
  FragU f;
  f.q[0] = *(const uint4*)p;
  f.q[1] = *(const uint4*)(p + 16);
  return f.v;
}

// B fragment 32x16 bf16 (B stored row-major as [N,K]): lane = N col (lane&15);
// same K layout as A: {0..7|8..15} then {16..23|24..31} by lane half. Two 16B loads.
__device__ __forceinline__ v16bf load_b_frag(const __bf16* base, int stride,
                                             int nrow, int kbase, int lane) {
  const __bf16* p = base + (size_t)nrow * stride + kbase + ((lane >> 4) << 3);
  FragU f;
  f.q[0] = *(const uint4*)p;
  f.q[1] = *(const uint4*)(p + 16);
  return f.v;
}

// ---------------------------------------------------------------------------
// Prep kernels (non-GEMM path)
// ---------------------------------------------------------------------------

// conv1: in [B,1,570,200] -> out [B,25,570,8]; kernel 1x49, stride (1,25), pad 24
__global__ void conv1_kernel(const float* __restrict__ x, const float* __restrict__ w,
                             const float* __restrict__ bia, float* __restrict__ h) {
  int i = blockIdx.x * blockDim.x + threadIdx.x;
  if (i >= B_ * 25 * T_ * 8) return;
  int ww = i & 7;
  int t  = (i >> 3) % T_;
  int oc = (i / (8 * T_)) % 25;
  int b  = i / (8 * T_ * 25);
  const float* xp = x + ((size_t)b * T_ + t) * PS_;
  const float* wp = w + oc * 49;
  int s0 = ww * 25 - 24;
  float acc = bia[oc];
#pragma unroll 7
  for (int k = 0; k < 49; ++k) {
    int s = s0 + k;
    if (s >= 0 && s < PS_) acc += xp[s] * wp[k];
  }
  h[i] = acc;
}

// 1x3 conv, 25->25 channels, pad 1 on width-8 axis (used for conv2 and conv3)
__global__ void conv3x_kernel(const float* __restrict__ in, const float* __restrict__ w,
                              const float* __restrict__ bia, float* __restrict__ out) {
  int i = blockIdx.x * blockDim.x + threadIdx.x;
  if (i >= B_ * 25 * T_ * 8) return;
  int ww = i & 7;
  int t  = (i >> 3) % T_;
  int oc = (i / (8 * T_)) % 25;
  int b  = i / (8 * T_ * 25);
  float acc = bia[oc];
  for (int ic = 0; ic < 25; ++ic) {
    const float* ip = in + (((size_t)b * 25 + ic) * T_ + t) * 8;
    const float* wp = w + (oc * 25 + ic) * 3;
#pragma unroll
    for (int j = 0; j < 3; ++j) {
      int s = ww + j - 1;
      if (s >= 0 && s < 8) acc += ip[s] * wp[j];
    }
  }
  out[i] = acc;
}

// GroupNorm(5 groups over 25 ch, stats over [5ch x 570 x 8]) + exact GELU, in place
__global__ void gn_gelu_kernel(float* __restrict__ h, const float* __restrict__ scale,
                               const float* __restrict__ bias) {
  int b = blockIdx.x / 5, g = blockIdx.x % 5;
  const int N = 5 * T_ * 8;  // 22800 contiguous floats
  float* p = h + ((size_t)(b * 25 + g * 5)) * T_ * 8;
  float s = 0.f, s2 = 0.f;
  for (int i = threadIdx.x; i < N; i += blockDim.x) {
    float v = p[i];
    s += v; s2 += v * v;
  }
  __shared__ float rs[256], rq[256];
  rs[threadIdx.x] = s; rq[threadIdx.x] = s2;
  __syncthreads();
  for (int o = 128; o > 0; o >>= 1) {
    if (threadIdx.x < o) { rs[threadIdx.x] += rs[threadIdx.x + o]; rq[threadIdx.x] += rq[threadIdx.x + o]; }
    __syncthreads();
  }
  float mean = rs[0] / N;
  float var  = rq[0] / N - mean * mean;
  float rstd = rsqrtf(var + 1e-5f);
  for (int i = threadIdx.x; i < N; i += blockDim.x) {
    int c = g * 5 + i / (T_ * 8);
    float v = (p[i] - mean) * rstd * scale[c] + bias[c];
    p[i] = 0.5f * v * (1.0f + erff(v * 0.70710678118654752f));
  }
}

// patch_emb = transpose(h) + (|rfft(x)|/PS) @ spec_w.T + spec_b; one block/token
__global__ void emb_kernel(const float* __restrict__ x, const float* __restrict__ h,
                           const float* __restrict__ sw, const float* __restrict__ sb,
                           float* __restrict__ pe) {
  int bt = blockIdx.x;               // token id
  int b = bt / T_, t = bt % T_;
  __shared__ float px[PS_];
  __shared__ float r[NF_];
  const float* xp = x + (size_t)bt * PS_;
  for (int i = threadIdx.x; i < PS_; i += blockDim.x) px[i] = xp[i];
  __syncthreads();
  for (int f = threadIdx.x; f < NF_; f += blockDim.x) {
    float re = 0.f, im = 0.f;
    float w0 = -6.283185307179586f * (float)f / 200.0f;
    for (int s = 0; s < PS_; ++s) {
      float a = w0 * (float)s;
      re += px[s] * cosf(a);
      im += px[s] * sinf(a);
    }
    r[f] = sqrtf(re * re + im * im) * (1.0f / 200.0f);
  }
  __syncthreads();
  for (int d = threadIdx.x; d < DM_; d += blockDim.x) {
    int c = d >> 3, ww = d & 7;
    float v = h[(((size_t)b * 25 + c) * T_ + t) * 8 + ww];
    float acc = sb[d];
    const float* swp = sw + d * NF_;
    for (int f = 0; f < NF_; ++f) acc += r[f] * swp[f];
    pe[(size_t)bt * DM_ + d] = v + acc;
  }
}

// depthwise 19x7 pos-conv over (ch,pn), residual add, emit bf16 A matrix [NT,KP]
__global__ void pos_kernel(const float* __restrict__ pe, const float* __restrict__ pw,
                           const float* __restrict__ pb, __bf16* __restrict__ A) {
  int i = blockIdx.x * blockDim.x + threadIdx.x;
  if (i >= B_ * T_ * DM_) return;
  int d = i % DM_;
  int t = (i / DM_) % T_;
  int b = i / (DM_ * T_);
  int y = t / NP_P, xx = t % NP_P;
  float acc = pb[d];
  const float* wp = pw + d * 133;  // (19*7)
  for (int ky = 0; ky < 19; ++ky) {
    int yy = y + ky - 9;
    if (yy < 0 || yy >= CH_) continue;
#pragma unroll
    for (int kx = 0; kx < 7; ++kx) {
      int xc = xx + kx - 3;
      if (xc < 0 || xc >= NP_P) continue;
      acc += pe[((size_t)b * T_ + yy * NP_P + xc) * DM_ + d] * wp[ky * 7 + kx];
    }
  }
  float v = pe[(size_t)(b * T_ + t) * DM_ + d] + acc;
  A[(size_t)(b * T_ + t) * KP_ + d] = (__bf16)v;
}

// fp32 [R,C] -> bf16 [Rp,Cp] with zero padding (weights + codebook conversion)
__global__ void cvt_pad_kernel(const float* __restrict__ src, __bf16* __restrict__ dst,
                               int R, int C, int Rp, int Cp) {
  int i = blockIdx.x * blockDim.x + threadIdx.x;
  if (i >= Rp * Cp) return;
  int c = i % Cp, r = i / Cp;
  float v = (r < R && c < C) ? src[(size_t)r * C + c] : 0.0f;
  dst[i] = (__bf16)v;
}

// |codebook_k|^2 in fp32 (block per code)
__global__ void cb_norm2_kernel(const float* __restrict__ cb, float* __restrict__ out) {
  int code = blockIdx.x;
  const float* p = cb + (size_t)code * LLM_;
  float s = 0.f;
  for (int i = threadIdx.x; i < LLM_; i += 256) { float v = p[i]; s += v * v; }
  __shared__ float rs[256];
  rs[threadIdx.x] = s;
  __syncthreads();
  for (int o = 128; o > 0; o >>= 1) {
    if (threadIdx.x < o) rs[threadIdx.x] += rs[threadIdx.x + o];
    __syncthreads();
  }
  if (threadIdx.x == 0) out[code] = rs[0];
}

// ---------------------------------------------------------------------------
// WMMA GEMM kernels
// ---------------------------------------------------------------------------

// proj[NT,LLM] (bf16) = A[NT,KP] @ inp_w[LLM,KP]^T + inp_b ; wave per 16x16 tile
__global__ void proj_gemm_kernel(const __bf16* __restrict__ A, const __bf16* __restrict__ Bw,
                                 const float* __restrict__ bias, __bf16* __restrict__ P) {
  int lane = threadIdx.x & 31, wave = threadIdx.x >> 5;
  int m0 = blockIdx.y * 16;
  int n0 = (blockIdx.x * 8 + wave) * 16;
  int arow = m0 + (lane & 15);
  int brow = n0 + (lane & 15);
  v8f acc = {};
#pragma unroll
  for (int kb = 0; kb < KP_; kb += 32) {
    v16bf a = load_a_frag(A, KP_, arow, kb, lane);
    v16bf b = load_b_frag(Bw, KP_, brow, kb, lane);
    acc = __builtin_amdgcn_wmma_f32_16x16x32_bf16(false, a, false, b, (short)0, acc,
                                                  false, false);
  }
  int n = n0 + (lane & 15);
  float bv = bias[n];
  int mbase = m0 + ((lane >> 4) << 3);
#pragma unroll
  for (int r = 0; r < 8; ++r)
    P[(size_t)(mbase + r) * LLM_ + n] = (__bf16)(acc[r] + bv);
}

// Fused score GEMM + argmin: dist = |c|^2 - 2 * proj.c  (|proj|^2 is constant/row)
// Block = 16 tokens staged once into LDS; 8 waves sweep 256 code tiles, 4 code
// tiles register-blocked per wave so each LDS A-fragment feeds 4 WMMAs.
__global__ void vq_kernel(const __bf16* __restrict__ P, const __bf16* __restrict__ CB,
                          const float* __restrict__ cn2, int* __restrict__ idx) {
  extern __shared__ char smem[];
  __bf16* sA = (__bf16*)smem;                                   // 16 x ASTRIDE_
  float*  sd = (float*)(smem + (size_t)16 * ASTRIDE_ * 2);      // [8][16]
  int*    si = (int*)((char*)sd + 8 * 16 * 4);                  // [8][16]

  int lane = threadIdx.x & 31, wave = threadIdx.x >> 5;
  int m0 = blockIdx.x * 16;

  // Stage A tile [16 x 4096] bf16 -> LDS (16B chunks, coalesced, padded stride)
  for (int c = threadIdx.x; c < 16 * (LLM_ / 8); c += 256) {
    int row = c >> 9;              // 512 chunks of 8 elems per row
    int col = (c & 511) << 3;
    *(uint4*)(sA + row * ASTRIDE_ + col) =
        *(const uint4*)(P + ((size_t)(m0 + row)) * LLM_ + col);
  }
  __syncthreads();

  float best[8];
  int   bidx[8];
#pragma unroll
  for (int r = 0; r < 8; ++r) { best[r] = 3.4e38f; bidx[r] = 0x7fffffff; }

  int arow = lane & 15;
  for (int it = 0; it < 8; ++it) {
    int ct0 = wave * 32 + it * 4;          // this wave's 4 code tiles
    int ncol = (lane & 15);
    // Warm WGP cache for the next group's codebook rows (global_prefetch_b8)
    if (it < 7) {
      __builtin_prefetch(CB + ((size_t)(ct0 + 4) * 16 + ncol) * LLM_, 0, 1);
      __builtin_prefetch(CB + ((size_t)(ct0 + 6) * 16 + ncol) * LLM_, 0, 1);
    }
    v8f acc0 = {}, acc1 = {}, acc2 = {}, acc3 = {};
#pragma unroll 2
    for (int kb = 0; kb < LLM_; kb += 32) {
      v16bf a  = lds_a_frag(sA, arow, kb, lane);
      v16bf b0 = load_b_frag(CB, LLM_, (ct0 + 0) * 16 + ncol, kb, lane);
      v16bf b1 = load_b_frag(CB, LLM_, (ct0 + 1) * 16 + ncol, kb, lane);
      v16bf b2 = load_b_frag(CB, LLM_, (ct0 + 2) * 16 + ncol, kb, lane);
      v16bf b3 = load_b_frag(CB, LLM_, (ct0 + 3) * 16 + ncol, kb, lane);
      acc0 = __builtin_amdgcn_wmma_f32_16x16x32_bf16(false, a, false, b0, (short)0, acc0, false, false);
      acc1 = __builtin_amdgcn_wmma_f32_16x16x32_bf16(false, a, false, b1, (short)0, acc1, false, false);
      acc2 = __builtin_amdgcn_wmma_f32_16x16x32_bf16(false, a, false, b2, (short)0, acc2, false, false);
      acc3 = __builtin_amdgcn_wmma_f32_16x16x32_bf16(false, a, false, b3, (short)0, acc3, false, false);
    }
#pragma unroll
    for (int j = 0; j < 4; ++j) {
      v8f acc = (j == 0) ? acc0 : (j == 1) ? acc1 : (j == 2) ? acc2 : acc3;
      int code = (ct0 + j) * 16 + ncol;
      float cn = cn2[code];
#pragma unroll
      for (int r = 0; r < 8; ++r) {
        float d = cn - 2.0f * acc[r];
        if (d < best[r] || (d == best[r] && code < bidx[r])) { best[r] = d; bidx[r] = code; }
      }
    }
  }

  // reduce across the 16 lanes of each half-wave (each row lives in one half)
  for (int off = 1; off < 16; off <<= 1) {
#pragma unroll
    for (int r = 0; r < 8; ++r) {
      float od = __shfl_xor(best[r], off, 32);
      int   oi = __shfl_xor(bidx[r], off, 32);
      if (od < best[r] || (od == best[r] && oi < bidx[r])) { best[r] = od; bidx[r] = oi; }
    }
  }
  if ((lane & 15) == 0) {
    int half = lane >> 4;  // rows 0..7 vs 8..15
#pragma unroll
    for (int r = 0; r < 8; ++r) {
      sd[wave * 16 + half * 8 + r] = best[r];
      si[wave * 16 + half * 8 + r] = bidx[r];
    }
  }
  __syncthreads();
  if (threadIdx.x < 16) {
    float bd = sd[threadIdx.x];
    int   bi = si[threadIdx.x];
    for (int w2 = 1; w2 < 8; ++w2) {
      float od = sd[w2 * 16 + threadIdx.x];
      int   oi = si[w2 * 16 + threadIdx.x];
      if (od < bd || (od == bd && oi < bi)) { bd = od; bi = oi; }
    }
    idx[m0 + threadIdx.x] = bi;
  }
}

// out[NT,DM] (fp32) = codebook[idx] @ outp_w^T + outp_b ; A rows gathered by idx
__global__ void out_gemm_kernel(const int* __restrict__ idx, const __bf16* __restrict__ CB,
                                const __bf16* __restrict__ Wo, const float* __restrict__ bias,
                                float* __restrict__ out) {
  int lane = threadIdx.x & 31, wave = threadIdx.x >> 5;
  int ntile = blockIdx.x * 8 + wave;
  if (ntile >= NO_ / 16) return;  // uniform per wave -> EXEC all-1 for WMMA waves
  int m0 = blockIdx.y * 16;
  int arow = idx[m0 + (lane & 15)];
  int brow = ntile * 16 + (lane & 15);
  v8f acc = {};
#pragma unroll 4
  for (int kb = 0; kb < LLM_; kb += 32) {
    v16bf a = load_a_frag(CB, LLM_, arow, kb, lane);
    v16bf b = load_b_frag(Wo, LLM_, brow, kb, lane);
    acc = __builtin_amdgcn_wmma_f32_16x16x32_bf16(false, a, false, b, (short)0, acc,
                                                  false, false);
  }
  int n = ntile * 16 + (lane & 15);
  if (n < DM_) {
    float bv = bias[n];
    int mbase = m0 + ((lane >> 4) << 3);
#pragma unroll
    for (int r = 0; r < 8; ++r)
      out[(size_t)(mbase + r) * DM_ + n] = acc[r] + bv;
  }
}

// ---------------------------------------------------------------------------
// Launch
// ---------------------------------------------------------------------------
extern "C" void kernel_launch(void* const* d_in, const int* in_sizes, int n_in,
                              void* d_out, int out_size, void* d_ws, size_t ws_size,
                              hipStream_t stream) {
  (void)in_sizes; (void)n_in; (void)out_size; (void)ws_size;
  const float* x      = (const float*)d_in[0];
  const float* c1w    = (const float*)d_in[1];
  const float* c1b    = (const float*)d_in[2];
  const float* gn1s   = (const float*)d_in[3];
  const float* gn1b   = (const float*)d_in[4];
  const float* c2w    = (const float*)d_in[5];
  const float* c2b    = (const float*)d_in[6];
  const float* gn2s   = (const float*)d_in[7];
  const float* gn2b   = (const float*)d_in[8];
  const float* c3w    = (const float*)d_in[9];
  const float* c3b    = (const float*)d_in[10];
  const float* gn3s   = (const float*)d_in[11];
  const float* gn3b   = (const float*)d_in[12];
  const float* spec_w = (const float*)d_in[13];
  const float* spec_b = (const float*)d_in[14];
  const float* pos_w  = (const float*)d_in[15];
  const float* pos_b  = (const float*)d_in[16];
  const float* inp_w  = (const float*)d_in[17];
  const float* inp_b  = (const float*)d_in[18];
  const float* cbk    = (const float*)d_in[19];
  const float* outp_w = (const float*)d_in[20];
  const float* outp_b = (const float*)d_in[21];
  float* out = (float*)d_out;

  // Workspace layout (all sizes multiples of 256 bytes)
  char* w = (char*)d_ws;
  const size_t SZ_H    = (size_t)B_ * 25 * T_ * 8 * 4;   // 14.59 MB
  const size_t SZ_PE   = (size_t)NT_ * DM_ * 4;          // 14.59 MB
  const size_t SZ_A    = (size_t)NT_ * KP_ * 2;          //  8.17 MB
  const size_t SZ_WI   = (size_t)LLM_ * KP_ * 2;         //  1.84 MB
  const size_t SZ_PROJ = (size_t)NT_ * LLM_ * 2;         // 149.4 MB
  const size_t SZ_CB   = (size_t)KC_ * LLM_ * 2;         // 33.55 MB
  const size_t SZ_CN   = (size_t)KC_ * 4;
  const size_t SZ_WO   = (size_t)NO_ * LLM_ * 2;         //  1.70 MB
  float*  h1      = (float*)w;                    w += SZ_H;
  float*  h2      = (float*)w;                    w += SZ_H;
  float*  pe      = (float*)w;                    w += SZ_PE;
  __bf16* A_bf    = (__bf16*)w;                   w += SZ_A;
  __bf16* Wi_bf   = (__bf16*)w;                   w += SZ_WI;
  __bf16* proj_bf = (__bf16*)w;                   w += SZ_PROJ;
  __bf16* cb_bf   = (__bf16*)w;                   w += SZ_CB;
  float*  cbn2    = (float*)w;                    w += SZ_CN;
  __bf16* Wo_bf   = (__bf16*)w;                   w += SZ_WO;
  int*    idx     = (int*)w;

  const int NEL = B_ * 25 * T_ * 8;  // 3,648,000

  // Weight conversions (independent of activation chain)
  cvt_pad_kernel<<<(LLM_ * KP_ + 255) / 256, 256, 0, stream>>>(inp_w, Wi_bf, LLM_, DM_, LLM_, KP_);
  cvt_pad_kernel<<<(KC_ * LLM_ + 255) / 256, 256, 0, stream>>>(cbk, cb_bf, KC_, LLM_, KC_, LLM_);
  cvt_pad_kernel<<<(NO_ * LLM_ + 255) / 256, 256, 0, stream>>>(outp_w, Wo_bf, DM_, LLM_, NO_, LLM_);
  cb_norm2_kernel<<<KC_, 256, 0, stream>>>(cbk, cbn2);

  // Zero A so the KP padding columns are 0 every call
  hipMemsetAsync(A_bf, 0, SZ_A, stream);

  // Conv / GN / GELU chain
  conv1_kernel<<<(NEL + 255) / 256, 256, 0, stream>>>(x, c1w, c1b, h1);
  gn_gelu_kernel<<<B_ * 5, 256, 0, stream>>>(h1, gn1s, gn1b);
  conv3x_kernel<<<(NEL + 255) / 256, 256, 0, stream>>>(h1, c2w, c2b, h2);
  gn_gelu_kernel<<<B_ * 5, 256, 0, stream>>>(h2, gn2s, gn2b);
  conv3x_kernel<<<(NEL + 255) / 256, 256, 0, stream>>>(h2, c3w, c3b, h1);
  gn_gelu_kernel<<<B_ * 5, 256, 0, stream>>>(h1, gn3s, gn3b);

  // Transpose + spectral branch, then depthwise pos conv -> bf16 A matrix
  emb_kernel<<<NT_, 256, 0, stream>>>(x, h1, spec_w, spec_b, pe);
  pos_kernel<<<(NT_ * DM_ + 255) / 256, 256, 0, stream>>>(pe, pos_w, pos_b, A_bf);

  // GEMM 1: proj (bf16 out, bias fused)
  proj_gemm_kernel<<<dim3(LLM_ / 16 / 8, NT_ / 16), 256, 0, stream>>>(A_bf, Wi_bf, inp_b, proj_bf);
  // GEMM 2 + fused argmin over 4096 codes (A tile LDS-staged, 132 KB dyn smem)
  size_t vq_smem = (size_t)16 * ASTRIDE_ * 2 + 8 * 16 * 4 * 2;
  vq_kernel<<<NT_ / 16, 256, vq_smem, stream>>>(proj_bf, cb_bf, cbn2, idx);
  // GEMM 3: gather codebook rows by idx, project back to DM, fp32 out
  out_gemm_kernel<<<dim3(2, NT_ / 16), 256, 0, stream>>>(idx, cb_bf, Wo_bf, outp_b, out);
}
